// Sum_Attention_3951369912994
// MI455X (gfx1250) — matrix operand order repaired and checked by reading, not compiled, not run on hardware
//
#include <hip/hip_runtime.h>

// ---------------- types ----------------
typedef __attribute__((ext_vector_type(16))) __bf16 v16bf;
typedef __attribute__((ext_vector_type(8)))  __bf16 v8bf;
typedef __attribute__((ext_vector_type(4)))  __bf16 v4bf;
typedef __attribute__((ext_vector_type(8)))  float  v8f;
typedef __attribute__((ext_vector_type(4)))  float  v4f;

#define B_   8
#define T_   243
#define J_   17
#define C_   512
#define H_   8
#define HD_  64
#define NTOK 33048          // B*T*J
#define SCALE_ 0.125f       // 1/sqrt(64)

static __device__ inline v8f wmma_bf16(v16bf a, v16bf b, v8f c) {
  return __builtin_amdgcn_wmma_f32_16x16x32_bf16(false, a, false, b, (short)0, c,
                                                 false, false);
}

static __device__ inline v16bf zero16() {
  v16bf z;
#pragma unroll
  for (int i = 0; i < 16; ++i) z[i] = (__bf16)0.0f;
  return z;
}
static __device__ inline v8bf zero8b() {
  v8bf z;
#pragma unroll
  for (int i = 0; i < 8; ++i) z[i] = (__bf16)0.0f;
  return z;
}

// A-matrix fragment (16x32 bf16, M x K). tile points at row0,k0; ld in elements.
// lane&15 = M row; (lane>>4)*8 = kh. elems 0..7 <-> k=kh..kh+7, 8..15 <-> k=16+kh..
static __device__ inline v16bf load_a_frag(const __bf16* tile, int ld, int lane) {
  int m  = lane & 15;
  int kh = (lane >> 4) << 3;
  const __bf16* p = tile + m * ld + kh;
  v8bf lo = *(const v8bf*)(p);
  v8bf hi = *(const v8bf*)(p + 16);
  v16bf r;
#pragma unroll
  for (int i = 0; i < 8; ++i) { r[i] = lo[i]; r[8 + i] = hi[i]; }
  return r;
}

// B-matrix fragment (32x16 bf16, K x N) from N-major storage tile[n][k] (ld elems).
// lane&15 = column n; kh=(lane>>4)*8. elems 0..7 <-> k=kh..kh+7, 8..15 <-> k=16+kh..
static __device__ inline v16bf load_b_frag(const __bf16* tile, int ld, int lane) {
  int n  = lane & 15;
  int kb = (lane >> 4) << 3;
  const __bf16* p = tile + n * ld + kb;
  v8bf lo = *(const v8bf*)(p);
  v8bf hi = *(const v8bf*)(p + 16);
  v16bf r;
#pragma unroll
  for (int i = 0; i < 8; ++i) { r[i] = lo[i]; r[8 + i] = hi[i]; }
  return r;
}

// ============================================================================
// Kernel 1: qkv = x @ w_qkv^T  -> scatter into Q/K/V [B,H,J,T,64] bf16
// block 256 thr (8 waves), tile 128(M) x 128(D), K-chunks of 32
// ============================================================================
__global__ __launch_bounds__(256) void qkv_gemm(const float* __restrict__ x,
                                                const float* __restrict__ wqkv,
                                                __bf16* __restrict__ Qb,
                                                __bf16* __restrict__ Kb,
                                                __bf16* __restrict__ Vb) {
  __shared__ __bf16 As[128 * 32];
  __shared__ __bf16 Bs[128 * 32];
  const int tid = threadIdx.x, lane = tid & 31, wid = tid >> 5;
  const int wm = wid & 1, wn = wid >> 1;      // 2 x 4 wave grid: 64 x 32 per wave
  const int m_base = blockIdx.x * 128;
  const int n_base = blockIdx.y * 128;

  v8f acc[4][2];
#pragma unroll
  for (int mi = 0; mi < 4; ++mi)
#pragma unroll
    for (int ni = 0; ni < 2; ++ni) acc[mi][ni] = (v8f){};

  for (int k0 = 0; k0 < C_; k0 += 32) {
    // ---- branch-free wide staging: 1024 float4 chunks, 4 per thread ----
#pragma unroll
    for (int u = 0; u < 4; ++u) {
      int e = (tid + u * 256) << 2;          // element index 0..4092
      int r = e >> 5, c = e & 31;
      int gm = m_base + r;
      int gsafe = (gm < NTOK) ? gm : (NTOK - 1);
      v4f va = *(const v4f*)(x + (size_t)gsafe * C_ + k0 + c);
      v4f vb = *(const v4f*)(wqkv + (size_t)(n_base + r) * C_ + k0 + c);
      bool ok = (gm < NTOK);
      v4bf pa, pb;
#pragma unroll
      for (int q = 0; q < 4; ++q) {
        pa[q] = (__bf16)(ok ? va[q] : 0.0f);
        pb[q] = (__bf16)vb[q];
      }
      *(v4bf*)(As + e) = pa;
      *(v4bf*)(Bs + e) = pb;
    }
    __syncthreads();
    v16bf a[4], bfr[2];
#pragma unroll
    for (int mi = 0; mi < 4; ++mi)
      a[mi] = load_a_frag(As + (wm * 64 + mi * 16) * 32, 32, lane);
#pragma unroll
    for (int ni = 0; ni < 2; ++ni)
      bfr[ni] = load_b_frag(Bs + (wn * 32 + ni * 16) * 32, 32, lane);
#pragma unroll
    for (int mi = 0; mi < 4; ++mi)
#pragma unroll
      for (int ni = 0; ni < 2; ++ni)
        acc[mi][ni] = wmma_bf16(a[mi], bfr[ni], acc[mi][ni]);
    __syncthreads();
  }

  const int col_l = lane & 15, row_l = (lane >> 4) * 8;
#pragma unroll
  for (int mi = 0; mi < 4; ++mi) {
#pragma unroll
    for (int ni = 0; ni < 2; ++ni) {
      int dg = n_base + wn * 32 + ni * 16 + col_l;       // 0..1535
      int which = dg >> 9;                               // 0:q 1:k 2:v
      int h = (dg >> 6) & 7, cc = dg & 63;
      __bf16* dst = (which == 0) ? Qb : (which == 1) ? Kb : Vb;
#pragma unroll
      for (int r = 0; r < 8; ++r) {
        int gm = m_base + wm * 64 + mi * 16 + row_l + r;
        if (gm < NTOK) {
          int j = gm % J_;
          int bt = gm / J_;
          int t = bt % T_;
          int bb = bt / T_;
          size_t off = (((size_t)((bb * H_ + h) * J_ + j)) * T_ + t) * HD_ + cc;
          dst[off] = (__bf16)acc[mi][ni][r];
        }
      }
    }
  }
}

// ============================================================================
// Kernel 2: per-(b,h,j) attention.  block = 128 thr (4 waves).
// S = Q K^T * scale, softmax, blend with att_map, O = P V  -> out_pre bf16 [N,512]
// Dynamic LDS: K[256][64] + Vt[64][256] + 4 x P[16][264]  ~= 97 KB
// ============================================================================
#define PS_LD 264
__global__ __launch_bounds__(128) void attn_kernel(const __bf16* __restrict__ Qb,
                                                   const __bf16* __restrict__ Kb,
                                                   const __bf16* __restrict__ Vb,
                                                   const float* __restrict__ att_map,
                                                   const float* __restrict__ weight,
                                                   __bf16* __restrict__ out_pre) {
  extern __shared__ __bf16 smem[];
  __bf16* Ks  = smem;                 // [256][64]   (rows >=243 zero)
  __bf16* Vts = Ks + 256 * 64;        // [64][256]   (cols >=243 zero)
  __bf16* Ps  = Vts + 64 * 256;       // [4][16][PS_LD]

  const int tid = threadIdx.x, lane = tid & 31, w = tid >> 5;
  const int bhj = blockIdx.x;                 // == (b*H + h)*J + j
  const int j = bhj % J_;
  const int h = (bhj / J_) % H_;
  const int b = bhj / (J_ * H_);
  const size_t head = (size_t)bhj * T_ * HD_;
  const __bf16* Qg = Qb + head;
  const __bf16* Kg = Kb + head;
  const __bf16* Vg = Vb + head;
  const float* attg = att_map + (size_t)bhj * T_ * T_;
  const float wgt = weight[0];

  // ---- stage K [s][c] with wide branch-free loads ----
#pragma unroll
  for (int u = 0; u < 16; ++u) {
    int e = (tid + u * 128) << 3;               // 0..16376, 8 elems each
    int s = e >> 6, c = e & 63;
    int ssafe = (s < T_) ? s : 0;
    v8bf kv = *(const v8bf*)(Kg + ssafe * HD_ + c);
    if (s >= T_) kv = zero8b();
    *(v8bf*)(Ks + e) = kv;
  }
  // ---- stage V transposed [c][s]: wide global reads, scatter LDS stores ----
#pragma unroll
  for (int u = 0; u < 16; ++u) {
    int e = (tid + u * 128) << 3;
    int s = e >> 6, c = e & 63;
    int ssafe = (s < T_) ? s : 0;
    v8bf vv = *(const v8bf*)(Vg + ssafe * HD_ + c);
    if (s >= T_) vv = zero8b();
#pragma unroll
    for (int q = 0; q < 8; ++q) Vts[(c + q) * 256 + s] = vv[q];
  }
  __syncthreads();

  __bf16* Pw = Ps + w * (16 * PS_LD);
  const int col_l = lane & 15, row_l = (lane >> 4) * 8;

  for (int rt = w; rt < 16; rt += 4) {
    const int m0 = rt * 16;
    // ---- Q A-fragments straight from global (rows >= T_ -> zero) ----
    v16bf aq0, aq1;
    {
      int m = m0 + col_l;
      int msafe = (m < T_) ? m : 0;
      int kh = (lane >> 4) << 3;
      const __bf16* p = Qg + msafe * HD_ + kh;
      v8bf a0 = *(const v8bf*)(p);
      v8bf a1 = *(const v8bf*)(p + 16);
      v8bf a2 = *(const v8bf*)(p + 32);
      v8bf a3 = *(const v8bf*)(p + 48);
      if (m >= T_) { a0 = zero8b(); a1 = zero8b(); a2 = zero8b(); a3 = zero8b(); }
#pragma unroll
      for (int i = 0; i < 8; ++i) {
        aq0[i] = a0[i]; aq0[8 + i] = a1[i];
        aq1[i] = a2[i]; aq1[8 + i] = a3[i];
      }
    }

    // ---- S strip: 16 tiles of 16x16 over key axis (padded to 256) ----
    v8f S[16];
#pragma unroll
    for (int sc = 0; sc < 16; ++sc) {
      v16bf bk0 = load_b_frag(Ks + sc * 16 * HD_, HD_, lane);
      v16bf bk1 = load_b_frag(Ks + sc * 16 * HD_ + 32, HD_, lane);
      v8f acc = (v8f){};
      acc = wmma_bf16(aq0, bk0, acc);
      acc = wmma_bf16(aq1, bk1, acc);
      int scol = sc * 16 + col_l;
#pragma unroll
      for (int r = 0; r < 8; ++r)
        acc[r] = (scol < T_) ? acc[r] * SCALE_ : -1e30f;
      S[sc] = acc;
    }

    // ---- softmax over key axis (rows live across 16-lane halves) ----
    v8f rmax = S[0];
#pragma unroll
    for (int sc = 1; sc < 16; ++sc)
#pragma unroll
      for (int r = 0; r < 8; ++r) rmax[r] = fmaxf(rmax[r], S[sc][r]);
#pragma unroll
    for (int off = 1; off < 16; off <<= 1)
#pragma unroll
      for (int r = 0; r < 8; ++r)
        rmax[r] = fmaxf(rmax[r], __shfl_xor(rmax[r], off, 32));

    v8f rsum = (v8f){};
#pragma unroll
    for (int sc = 0; sc < 16; ++sc)
#pragma unroll
      for (int r = 0; r < 8; ++r) {
        float e = __expf(S[sc][r] - rmax[r]);
        S[sc][r] = e;
        rsum[r] += e;
      }
#pragma unroll
    for (int off = 1; off < 16; off <<= 1)
#pragma unroll
      for (int r = 0; r < 8; ++r) rsum[r] += __shfl_xor(rsum[r], off, 32);

    v8f rinv;
#pragma unroll
    for (int r = 0; r < 8; ++r) rinv[r] = (rsum[r] > 0.f) ? 1.f / rsum[r] : 0.f;

    // ---- blend with att_map, write P strip (bf16) to per-wave LDS ----
#pragma unroll
    for (int sc = 0; sc < 16; ++sc) {
      int scol = sc * 16 + col_l;
#pragma unroll
      for (int r = 0; r < 8; ++r) {
        int t = m0 + row_l + r;
        float p = 0.0f;
        if (scol < T_) {
          p = wgt * S[sc][r] * rinv[r];
          if (t < T_) p += (1.0f - wgt) * attg[(size_t)t * T_ + scol];
        }
        Pw[(row_l + r) * PS_LD + scol] = (__bf16)p;
      }
    }

    // ---- O = P @ V : 4 output col-tiles (hd=64), k-chunks of 32 over s ----
    v8f oacc[4];
#pragma unroll
    for (int nt = 0; nt < 4; ++nt) oacc[nt] = (v8f){};
#pragma unroll
    for (int kc = 0; kc < 8; ++kc) {
      v16bf pa = load_a_frag(Pw + kc * 32, PS_LD, lane);
#pragma unroll
      for (int nt = 0; nt < 4; ++nt) {
        v16bf bv = load_b_frag(Vts + nt * 16 * 256 + kc * 32, 256, lane);
        oacc[nt] = wmma_bf16(pa, bv, oacc[nt]);
      }
    }
#pragma unroll
    for (int nt = 0; nt < 4; ++nt) {
      int cc = nt * 16 + col_l;
#pragma unroll
      for (int r = 0; r < 8; ++r) {
        int t = m0 + row_l + r;
        if (t < T_) {
          size_t n = ((size_t)(b * T_ + t)) * J_ + j;     // token row
          out_pre[n * C_ + h * HD_ + cc] = (__bf16)oacc[nt][r];
        }
      }
    }
  }
}

// ============================================================================
// Kernel 3: out = out_pre @ w_proj^T + b_proj  (fp32 output)
// ============================================================================
__global__ __launch_bounds__(256) void proj_gemm(const __bf16* __restrict__ pre,
                                                 const float* __restrict__ wproj,
                                                 const float* __restrict__ bias,
                                                 float* __restrict__ out) {
  __shared__ __bf16 As[128 * 32];
  __shared__ __bf16 Bs[128 * 32];
  const int tid = threadIdx.x, lane = tid & 31, wid = tid >> 5;
  const int wm = wid & 1, wn = wid >> 1;
  const int m_base = blockIdx.x * 128;
  const int n_base = blockIdx.y * 128;

  v8f acc[4][2];
#pragma unroll
  for (int mi = 0; mi < 4; ++mi)
#pragma unroll
    for (int ni = 0; ni < 2; ++ni) acc[mi][ni] = (v8f){};

  for (int k0 = 0; k0 < C_; k0 += 32) {
    // A: bf16 rows, 16-byte chunks, branch-free
#pragma unroll
    for (int u = 0; u < 2; ++u) {
      int e = (tid + u * 256) << 3;          // 8 elems each, 0..4088
      int r = e >> 5, c = e & 31;
      int gm = m_base + r;
      int gsafe = (gm < NTOK) ? gm : (NTOK - 1);
      v8bf a = *(const v8bf*)(pre + (size_t)gsafe * C_ + k0 + c);
      if (gm >= NTOK) a = zero8b();
      *(v8bf*)(As + e) = a;
    }
    // B: fp32 -> bf16, float4 chunks
#pragma unroll
    for (int u = 0; u < 4; ++u) {
      int e = (tid + u * 256) << 2;
      int r = e >> 5, c = e & 31;
      v4f vb = *(const v4f*)(wproj + (size_t)(n_base + r) * C_ + k0 + c);
      v4bf pb;
#pragma unroll
      for (int q = 0; q < 4; ++q) pb[q] = (__bf16)vb[q];
      *(v4bf*)(Bs + e) = pb;
    }
    __syncthreads();
    v16bf a[4], bfr[2];
#pragma unroll
    for (int mi = 0; mi < 4; ++mi)
      a[mi] = load_a_frag(As + (wm * 64 + mi * 16) * 32, 32, lane);
#pragma unroll
    for (int ni = 0; ni < 2; ++ni)
      bfr[ni] = load_b_frag(Bs + (wn * 32 + ni * 16) * 32, 32, lane);
#pragma unroll
    for (int mi = 0; mi < 4; ++mi)
#pragma unroll
      for (int ni = 0; ni < 2; ++ni)
        acc[mi][ni] = wmma_bf16(a[mi], bfr[ni], acc[mi][ni]);
    __syncthreads();
  }

  const int col_l = lane & 15, row_l = (lane >> 4) * 8;
#pragma unroll
  for (int mi = 0; mi < 4; ++mi) {
#pragma unroll
    for (int ni = 0; ni < 2; ++ni) {
      int cg = n_base + wn * 32 + ni * 16 + col_l;
      float bv = bias[cg];
#pragma unroll
      for (int r = 0; r < 8; ++r) {
        int gm = m_base + wm * 64 + mi * 16 + row_l + r;
        if (gm < NTOK) out[(size_t)gm * C_ + cg] = acc[mi][ni][r] + bv;
      }
    }
  }
}

// ============================================================================
extern "C" void kernel_launch(void* const* d_in, const int* in_sizes, int n_in,
                              void* d_out, int out_size, void* d_ws, size_t ws_size,
                              hipStream_t stream) {
  const float* x      = (const float*)d_in[0];
  const float* attmap = (const float*)d_in[1];
  const float* weight = (const float*)d_in[2];
  const float* w_qkv  = (const float*)d_in[3];
  const float* w_proj = (const float*)d_in[4];
  const float* b_proj = (const float*)d_in[5];
  float* out = (float*)d_out;

  const size_t head_elems = (size_t)B_ * H_ * J_ * T_ * HD_;  // 16,920,576
  __bf16* Qb  = (__bf16*)d_ws;
  __bf16* Kb  = Qb + head_elems;
  __bf16* Vb  = Kb + head_elems;
  __bf16* Pre = Vb + head_elems;                              // [NTOK, 512] bf16

  (void)in_sizes; (void)n_in; (void)out_size; (void)ws_size;

  dim3 g1((NTOK + 127) / 128, (3 * C_) / 128);   // 259 x 12
  qkv_gemm<<<g1, 256, 0, stream>>>(x, w_qkv, Qb, Kb, Vb);

  const size_t attn_lds = (256 * 64 + 64 * 256 + 4 * 16 * PS_LD) * sizeof(__bf16);
  attn_kernel<<<B_ * H_ * J_, 128, attn_lds, stream>>>(Qb, Kb, Vb, attmap, weight,
                                                       Pre);

  dim3 g3((NTOK + 127) / 128, C_ / 128);         // 259 x 4
  proj_gemm<<<g3, 256, 0, stream>>>(Pre, w_proj, b_proj, out);
}
